// GAT_9732395892851
// MI455X (gfx1250) — matrix operand order repaired and checked by reading, not compiled, not run on hardware
//
#include <hip/hip_runtime.h>
#include <hip/hip_bf16.h>
#include <float.h>

// ---------- WMMA vector types (CDNA5 / gfx1250, wave32) ----------
typedef __attribute__((ext_vector_type(16))) __bf16 v16bf;
typedef __attribute__((ext_vector_type(8)))  __bf16 v8bf;
typedef __attribute__((ext_vector_type(8)))  float  v8f;

#define NEG_SLOPE 0.2f

__device__ __forceinline__ float waveReduceAdd(float v) {
#pragma unroll
  for (int off = 16; off > 0; off >>= 1) v += __shfl_xor(v, off, 32);
  return v;
}

__device__ __forceinline__ void atomicMaxFloat(float* addr, float val) {
  if (val >= 0.f) atomicMax((int*)addr, __float_as_int(val));
  else            atomicMin((unsigned int*)addr, (unsigned int)__float_as_int(val));
}

// ---------- generic fill ----------
__global__ void GAT_fill_f32(float* p, float v, long n) {
  long gid = (long)blockIdx.x * blockDim.x + threadIdx.x;
  if (gid < n) p[gid] = v;
}

// ---------- f32 -> bf16 row-major convert ----------
__global__ void GAT_cvt_bf16(const float* __restrict__ in, __bf16* __restrict__ out, long n) {
  long gid = (long)blockIdx.x * blockDim.x + threadIdx.x;
  if (gid < n) out[gid] = (__bf16)in[gid];
}

// ---------- pack W [K, ncols] f32 into WMMA B-operand tiles (bf16) ----------
// Packed slot idx = ((coltile*kchunks + kc)*32 + lane)*16 + t
// lane 0..15  : col = coltile*16 + lane,      element t -> K = kc*32 + (t&7) + (t>=8 ? 16 : 0)
// lane 16..31 : col = coltile*16 + (lane-16), element t -> K = kc*32 + 8 + (t&7) + (t>=8 ? 16 : 0)
__global__ void GAT_pack_w(const float* __restrict__ W, __bf16* __restrict__ out,
                           int K, int ncols) {
  int idx = blockIdx.x * blockDim.x + threadIdx.x;
  int total = K * ncols;
  if (idx >= total) return;
  int t       = idx & 15;
  int lane    = (idx >> 4) & 31;
  int tile    = idx >> 9;
  int kchunks = K >> 5;
  int kc      = tile % kchunks;
  int coltile = tile / kchunks;
  int k   = kc * 32 + ((lane & 16) ? 8 : 0) + (t & 7) + ((t & 8) ? 16 : 0);
  int col = coltile * 16 + (lane & 15);
  out[idx] = (__bf16)W[k * ncols + col];
}

// ---------- bf16 WMMA GEMM: D[Nrows,ncols] = A[Nrows,K] * Bpacked ----------
// 4 waves/block; wave w owns 16x16 output tile (blockIdx.y*4 + w).
// A layout per ISA (16-bit A 16x32): lane<16 holds K kc*32+[0..7] and +[16..23],
// lane>=16 holds K kc*32+[8..15] and +[24..31]  -> two contiguous 16B loads.
__global__ void GAT_gemm_wmma(const __bf16* __restrict__ A, const __bf16* __restrict__ Bp,
                              float* __restrict__ D, int Nrows, int K, int ncols) {
  int wave = threadIdx.x >> 5;
  int lane = threadIdx.x & 31;
  int rowbase = blockIdx.x * 16;
  int coltile = blockIdx.y * 4 + wave;
  if (coltile * 16 >= ncols) return;           // wave-uniform: EXEC stays all-ones
  int kchunks = K >> 5;
  int l15 = lane & 15;
  int row = rowbase + l15;
  if (row >= Nrows) row = Nrows - 1;           // clamp (keeps EXEC full for WMMA)
  const __bf16* arow = A + (long)row * K + ((lane & 16) ? 8 : 0);
  const v16bf*  bp   = (const v16bf*)Bp + (long)coltile * kchunks * 32 + lane;

  v8f c = {};
  for (int kc = 0; kc < kchunks; ++kc) {
    const __bf16* ap = arow + kc * 32;
    v8bf lo = *(const v8bf*)(ap);
    v8bf hi = *(const v8bf*)(ap + 16);
    v16bf a;
#pragma unroll
    for (int i = 0; i < 8; ++i) { a[i] = lo[i]; a[i + 8] = hi[i]; }
    v16bf b = bp[kc * 32];
    c = __builtin_amdgcn_wmma_f32_16x16x32_bf16(false, a, false, b, (short)0, c,
                                                false, false);
  }
  // D layout: VGPR r, lane<16 -> M=r ; lane>=16 -> M=r+8 ; N = lane&15
  int orow0 = rowbase + ((lane & 16) ? 8 : 0);
  int col   = coltile * 16 + l15;
#pragma unroll
  for (int r = 0; r < 8; ++r) {
    int orow = orow0 + r;
    if (orow < Nrows) D[(long)orow * ncols + col] = c[r];
  }
}

// ---------- per-(node,head) attention logits: wave-per-(n,h) ----------
__global__ void GAT_alpha(const float* __restrict__ h, const float* __restrict__ a_src,
                          const float* __restrict__ a_dst, float* __restrict__ as,
                          float* __restrict__ ad, int NH, int H, int C) {
  long gid = (long)blockIdx.x * blockDim.x + threadIdx.x;
  int wid = (int)(gid >> 5), lane = (int)(gid & 31);
  if (wid >= NH) return;
  int hh = wid % H;
  const float* hp = h + (long)wid * C;
  float ps = 0.f, pd = 0.f;
  for (int cc = lane; cc < C; cc += 32) {
    float v = hp[cc];
    ps += v * a_src[hh * C + cc];
    pd += v * a_dst[hh * C + cc];
  }
  ps = waveReduceAdd(ps);
  pd = waveReduceAdd(pd);
  if (lane == 0) { as[wid] = ps; ad[wid] = pd; }
}

// ---------- edge pass 1: segment max over dst ----------
__global__ void GAT_edge_max(const int* __restrict__ ei, int E, int N, int H,
                             const float* __restrict__ as, const float* __restrict__ ad,
                             float* __restrict__ emax) {
  long gid = (long)blockIdx.x * blockDim.x + threadIdx.x;
  long total = (long)(E + N) * H;
  if (gid >= total) return;
  int hh = (int)(gid % H);
  long e = gid / H;
  int s, d;
  if (e < E) { s = ei[e]; d = ei[(long)E + e]; } else { s = d = (int)(e - E); }
  float ev = as[(long)s * H + hh] + ad[(long)d * H + hh];
  ev = ev > 0.f ? ev : ev * NEG_SLOPE;
  atomicMaxFloat(&emax[(long)d * H + hh], ev);
}

// ---------- edge pass 2: exp + segment sum ----------
__global__ void GAT_edge_exp(const int* __restrict__ ei, int E, int N, int H,
                             const float* __restrict__ as, const float* __restrict__ ad,
                             const float* __restrict__ emax, float* __restrict__ ee,
                             float* __restrict__ denom) {
  long gid = (long)blockIdx.x * blockDim.x + threadIdx.x;
  long total = (long)(E + N) * H;
  if (gid >= total) return;
  int hh = (int)(gid % H);
  long e = gid / H;
  int s, d;
  if (e < E) { s = ei[e]; d = ei[(long)E + e]; } else { s = d = (int)(e - E); }
  float ev = as[(long)s * H + hh] + ad[(long)d * H + hh];
  ev = ev > 0.f ? ev : ev * NEG_SLOPE;
  float x = expf(ev - emax[(long)d * H + hh]);
  ee[gid] = x;
  atomicAdd(&denom[(long)d * H + hh], x);
}

// ---------- edge pass 3: scatter msg; wave-per-(edge,head), coalesced ----------
__global__ void GAT_edge_scatter(const int* __restrict__ ei, int E, int N, int H, int C,
                                 const float* __restrict__ ee, const float* __restrict__ denom,
                                 const float* __restrict__ h, float* __restrict__ acc) {
  long gid = (long)blockIdx.x * blockDim.x + threadIdx.x;
  long wid = gid >> 5;
  int lane = (int)(gid & 31);
  long total = (long)(E + N) * H;
  if (wid >= total) return;
  int hh = (int)(wid % H);
  long e = wid / H;
  int s, d;
  if (e < E) { s = ei[e]; d = ei[(long)E + e]; } else { s = d = (int)(e - E); }
  float a = ee[wid] / denom[(long)d * H + hh];
  const float* hp = h + ((long)s * H + hh) * C;
  float* op = acc + ((long)d * H + hh) * C;
  for (int cc = lane; cc < C; cc += 32) atomicAdd(&op[cc], hp[cc] * a);
}

// ---------- bias + ELU + convert to bf16 (layer-1 output -> layer-2 input) ----
__global__ void GAT_elu_bias_cvt(const float* __restrict__ acc, const float* __restrict__ bias,
                                 __bf16* __restrict__ out, long n, int cols) {
  long gid = (long)blockIdx.x * blockDim.x + threadIdx.x;
  if (gid >= n) return;
  float v = acc[gid] + bias[gid % cols];
  v = v > 0.f ? v : (expf(v) - 1.f);
  out[gid] = (__bf16)v;
}

// ---------- in-place bias add (layer-2 output) ----------
__global__ void GAT_bias_add(float* __restrict__ acc, const float* __restrict__ bias,
                             long n, int cols) {
  long gid = (long)blockIdx.x * blockDim.x + threadIdx.x;
  if (gid < n) acc[gid] += bias[gid % cols];
}

// ---------- decode: wave-per-pair dot product ----------
__global__ void GAT_decode(const float* __restrict__ z, const int* __restrict__ eli,
                           int EL, int C, float* __restrict__ out) {
  long gid = (long)blockIdx.x * blockDim.x + threadIdx.x;
  int wid = (int)(gid >> 5), lane = (int)(gid & 31);
  if (wid >= EL) return;
  int a = eli[wid], b = eli[(long)EL + wid];
  const float* za = z + (long)a * C;
  const float* zb = z + (long)b * C;
  float p = 0.f;
  for (int cc = lane; cc < C; cc += 32) p += za[cc] * zb[cc];
  p = waveReduceAdd(p);
  if (lane == 0) out[wid] = p;
}

// =========================== host launcher ===========================
static inline long cdiv(long a, long b) { return (a + b - 1) / b; }

extern "C" void kernel_launch(void* const* d_in, const int* in_sizes, int n_in,
                              void* d_out, int out_size, void* d_ws, size_t ws_size,
                              hipStream_t stream) {
  const float* x   = (const float*)d_in[0];
  const int*   ei  = (const int*)d_in[1];
  const int*   eli = (const int*)d_in[2];
  const float* W1  = (const float*)d_in[3];
  const float* as1 = (const float*)d_in[4];
  const float* ad1 = (const float*)d_in[5];
  const float* b1  = (const float*)d_in[6];
  const float* W2  = (const float*)d_in[7];
  const float* as2 = (const float*)d_in[8];
  const float* ad2 = (const float*)d_in[9];
  const float* b2  = (const float*)d_in[10];

  const int HC = in_sizes[6];          // H*C
  const int F  = in_sizes[3] / HC;     // F_in
  const int N  = in_sizes[0] / F;
  const int C  = in_sizes[10];
  const int H  = HC / C;
  const int E  = in_sizes[1] / 2;
  const int EL = in_sizes[2] / 2;
  const long Etot = (long)E + N;

  // ----- workspace layout (256B aligned), buffers reused across layers -----
  char* ws = (char*)d_ws;
  size_t off = 0;
  auto carve = [&](size_t bytes) -> char* {
    char* p = ws + off;
    off = (off + bytes + 255) & ~(size_t)255;
    return p;
  };
  int maxK = F > HC ? F : HC;
  __bf16* xzb   = (__bf16*)carve((size_t)N * maxK * sizeof(__bf16));       // x_bf16 then z_bf16
  __bf16* wpack = (__bf16*)carve((size_t)maxK * HC * sizeof(__bf16));      // packed W (per layer)
  float*  hbuf  = (float*)carve((size_t)N * HC * sizeof(float));           // h1 then h2
  float*  acc   = (float*)carve((size_t)N * HC * sizeof(float));           // out1 then out2
  float*  asb   = (float*)carve((size_t)N * H * sizeof(float));
  float*  adb   = (float*)carve((size_t)N * H * sizeof(float));
  float*  emax  = (float*)carve((size_t)N * H * sizeof(float));
  float*  denom = (float*)carve((size_t)N * H * sizeof(float));
  float*  ee    = (float*)carve((size_t)Etot * H * sizeof(float));

  const int TB = 256;

  // ================= Layer 1 (heads=H, channels=C, K=F) =================
  {
    long n = (long)N * F;
    GAT_cvt_bf16<<<cdiv(n, TB), TB, 0, stream>>>(x, xzb, n);
    GAT_pack_w<<<cdiv((long)F * HC, TB), TB, 0, stream>>>(W1, wpack, F, HC);
    dim3 g1((N + 15) / 16, HC / 64);
    GAT_gemm_wmma<<<g1, 128, 0, stream>>>(xzb, wpack, hbuf, N, F, HC);

    long NH = (long)N * H;
    GAT_alpha<<<cdiv(NH * 32, TB), TB, 0, stream>>>(hbuf, as1, ad1, asb, adb, (int)NH, H, C);
    GAT_fill_f32<<<cdiv(NH, TB), TB, 0, stream>>>(emax, -FLT_MAX, NH);
    GAT_fill_f32<<<cdiv(NH, TB), TB, 0, stream>>>(denom, 0.f, NH);
    GAT_fill_f32<<<cdiv((long)N * HC, TB), TB, 0, stream>>>(acc, 0.f, (long)N * HC);

    long tot = Etot * H;
    GAT_edge_max<<<cdiv(tot, TB), TB, 0, stream>>>(ei, E, N, H, asb, adb, emax);
    GAT_edge_exp<<<cdiv(tot, TB), TB, 0, stream>>>(ei, E, N, H, asb, adb, emax, ee, denom);
    GAT_edge_scatter<<<cdiv(tot * 32, TB), TB, 0, stream>>>(ei, E, N, H, C, ee, denom, hbuf, acc);

    long nz = (long)N * HC;
    GAT_elu_bias_cvt<<<cdiv(nz, TB), TB, 0, stream>>>(acc, b1, xzb, nz, HC);
  }

  // ================= Layer 2 (heads=1, channels=C, K=HC) ================
  {
    GAT_pack_w<<<cdiv((long)HC * C, TB), TB, 0, stream>>>(W2, wpack, HC, C);
    dim3 g2((N + 15) / 16, (C + 63) / 64);
    GAT_gemm_wmma<<<g2, 128, 0, stream>>>(xzb, wpack, hbuf, N, HC, C);

    long NH = (long)N;  // H2 = 1
    GAT_alpha<<<cdiv(NH * 32, TB), TB, 0, stream>>>(hbuf, as2, ad2, asb, adb, (int)NH, 1, C);
    GAT_fill_f32<<<cdiv(NH, TB), TB, 0, stream>>>(emax, -FLT_MAX, NH);
    GAT_fill_f32<<<cdiv(NH, TB), TB, 0, stream>>>(denom, 0.f, NH);
    GAT_fill_f32<<<cdiv((long)N * C, TB), TB, 0, stream>>>(acc, 0.f, (long)N * C);

    long tot = Etot;
    GAT_edge_max<<<cdiv(tot, TB), TB, 0, stream>>>(ei, E, N, 1, asb, adb, emax);
    GAT_edge_exp<<<cdiv(tot, TB), TB, 0, stream>>>(ei, E, N, 1, asb, adb, emax, ee, denom);
    GAT_edge_scatter<<<cdiv(tot * 32, TB), TB, 0, stream>>>(ei, E, N, 1, C, ee, denom, hbuf, acc);

    long nz = (long)N * C;
    GAT_bias_add<<<cdiv(nz, TB), TB, 0, stream>>>(acc, b2, nz, C);
  }

  // ================= decode =================
  GAT_decode<<<cdiv((long)EL * 32, TB), TB, 0, stream>>>(acc, eli, EL, C, (float*)d_out);
}
